// MoELayer_63900523430580
// MI455X (gfx1250) — compile-verified, matrix operand order checked by reading, not run on hardware
//
#include <hip/hip_runtime.h>
#include <hip/hip_bf16.h>

#define T_TOK   8192
#define DIM_D   1024
#define DIM_H   4096
#define NEXP    8

typedef __bf16 bf16_t;
typedef bf16_t v16bf __attribute__((ext_vector_type(16)));
typedef bf16_t v8bf  __attribute__((ext_vector_type(8)));
typedef float  v8f   __attribute__((ext_vector_type(8)));

__device__ __forceinline__ bf16_t f2bf(float f) { return (bf16_t)f; }

// A-fragment (16x32 bf16) from LDS, per ISA 7.12.2 16-bit A layout:
// lane m=lane&15 holds row m; K chunks [ (lane>>4)*8 , +8 ) and [ 16+(lane>>4)*8 , +8 ).
__device__ __forceinline__ v16bf lds_a_frag(const bf16_t* row, int koff) {
  const v8bf lo = *reinterpret_cast<const v8bf*>(row + koff);
  const v8bf hi = *reinterpret_cast<const v8bf*>(row + koff + 16);
  return __builtin_shufflevector(lo, hi, 0,1,2,3,4,5,6,7,8,9,10,11,12,13,14,15);
}

// B-fragment (32x16) from global fp32 weights (row-major [K, ldn]), cvt to bf16.
// lane n=lane&15 holds column n; same K chunking as A. Lanes 0..15 read
// consecutive columns at a fixed k -> 64B coalesced per half-wave.
__device__ __forceinline__ v16bf glb_b_frag(const float* __restrict__ w, int k0, int n, int ldn) {
  v16bf b;
  #pragma unroll
  for (int j = 0; j < 8; ++j) {
    b[j]     = f2bf(w[(k0 + j) * ldn + n]);
    b[j + 8] = f2bf(w[(k0 + 16 + j) * ldn + n]);
  }
  return b;
}

__device__ __forceinline__ v8f wmma_bf16(v16bf a, v16bf b, v8f c) {
  return __builtin_amdgcn_wmma_f32_16x16x32_bf16(false, a, false, b, (short)0, c, false, false);
}

// ---------------- Router: gate scores, top-2 softmax, dispatch lists ----------------
__global__ __launch_bounds__(256) void router_kernel(
    const float* __restrict__ x, const float* __restrict__ gw,
    int* __restrict__ counts, int* __restrict__ tok,
    float* __restrict__ wgt, float* __restrict__ usage)
{
  const int t    = blockIdx.x * 8 + (threadIdx.x >> 5);
  const int lane = threadIdx.x & 31;
  float acc[NEXP] = {};
  const float* xr = x + (size_t)t * DIM_D;
  for (int d = lane; d < DIM_D; d += 32) {
    float xv = xr[d];
    #pragma unroll
    for (int e = 0; e < NEXP; ++e) acc[e] += xv * gw[d * NEXP + e];
  }
  #pragma unroll
  for (int off = 16; off > 0; off >>= 1) {
    #pragma unroll
    for (int e = 0; e < NEXP; ++e) acc[e] += __shfl_xor(acc[e], off, 32);
  }
  if (lane == 0) {
    float mx = acc[0];
    #pragma unroll
    for (int e = 1; e < NEXP; ++e) mx = fmaxf(mx, acc[e]);
    float ex[NEXP], se = 0.f;
    #pragma unroll
    for (int e = 0; e < NEXP; ++e) { ex[e] = __expf(acc[e] - mx); se += ex[e]; }
    float inv = 1.f / se;
    #pragma unroll
    for (int e = 0; e < NEXP; ++e) atomicAdd(&usage[e], ex[e] * inv);
    int i1 = 0;
    #pragma unroll
    for (int e = 1; e < NEXP; ++e) if (acc[e] > acc[i1]) i1 = e;
    int i2 = (i1 == 0) ? 1 : 0;
    #pragma unroll
    for (int e = 0; e < NEXP; ++e) if (e != i1 && acc[e] > acc[i2]) i2 = e;
    // softmax over the two selected logits
    float p = 1.f / (1.f + __expf(acc[i2] - acc[i1]));
    int s1 = atomicAdd(&counts[i1], 1);
    tok[i1 * T_TOK + s1] = t; wgt[i1 * T_TOK + s1] = p;
    int s2 = atomicAdd(&counts[i2], 1);
    tok[i2 * T_TOK + s2] = t; wgt[i2 * T_TOK + s2] = 1.f - p;
  }
}

// ---------------- Fused per-expert SwiGLU + down-proj, bf16 WMMA, M=32 ----------------
__global__ __launch_bounds__(256, 1) void moe_expert_kernel(
    const float* __restrict__ x,  const float* __restrict__ w1,
    const float* __restrict__ w3, const float* __restrict__ w2,
    const int* __restrict__ counts, const int* __restrict__ tok,
    const float* __restrict__ wgt, float* __restrict__ out)
{
  __shared__ alignas(16) bf16_t xa[32 * DIM_D];   //  64 KB: gathered x tile, bf16
  __shared__ alignas(16) bf16_t hs[32 * DIM_H];   // 256 KB: full h tile, bf16 (320 KB WGP LDS total)

  const int e    = blockIdx.x >> 8;        // 256 tiles of 32 tokens per expert
  const int tile = blockIdx.x & 255;
  const int cnt  = counts[e];
  if (tile * 32 >= cnt) return;            // empty tile: whole block exits before WMMA

  const int*   tokl = tok + e * T_TOK + tile * 32;
  const float* wgtl = wgt + e * T_TOK + tile * 32;
  const int cnt_rem = cnt - tile * 32;     // rows >= cnt_rem are weight-0 padding

  for (int i = threadIdx.x; i < 32 * DIM_D; i += 256) {
    int m = i >> 10, d = i & (DIM_D - 1);
    int tid = tokl[m < cnt_rem ? m : 0];
    xa[i] = f2bf(x[(size_t)tid * DIM_D + d]);
  }
  __syncthreads();

  const int lane = threadIdx.x & 31;
  const int wv   = threadIdx.x >> 5;
  const int mrow = lane & 15;
  const int ncol = lane & 15;
  const int koff = (lane >> 4) << 3;
  const int rbase = (lane >> 4) << 3;      // C-tile row base for this lane

  const float* w1e = w1 + (size_t)e * DIM_D * DIM_H;
  const float* w3e = w3 + (size_t)e * DIM_D * DIM_H;
  const float* w2e = w2 + (size_t)e * DIM_H * DIM_D;

  // ---- Phase 1: h[32, 4096] = silu(x@w1) * (x@w3); each B frag feeds 2 WMMAs ----
  const bf16_t* arow0 = xa + mrow * DIM_D;
  const bf16_t* arow1 = xa + (16 + mrow) * DIM_D;
  for (int j = 0; j < 32; ++j) {
    const int col0 = (wv + 8 * j) * 16;
    v8f acc1a = {}, acc1b = {}, acc3a = {}, acc3b = {};
    for (int kk = 0; kk < DIM_D / 32; ++kk) {
      const int k0 = kk * 32 + koff;
      v16bf a0 = lds_a_frag(arow0 + kk * 32, koff);
      v16bf a1 = lds_a_frag(arow1 + kk * 32, koff);
      v16bf b1 = glb_b_frag(w1e, k0, col0 + ncol, DIM_H);
      v16bf b3 = glb_b_frag(w3e, k0, col0 + ncol, DIM_H);
      acc1a = wmma_bf16(a0, b1, acc1a);
      acc1b = wmma_bf16(a1, b1, acc1b);
      acc3a = wmma_bf16(a0, b3, acc3a);
      acc3b = wmma_bf16(a1, b3, acc3b);
    }
    #pragma unroll
    for (int r = 0; r < 8; ++r) {           // C layout: lane n=lane&15, row = r + rbase
      const int m = r + rbase;
      float g0 = acc1a[r], u0 = acc3a[r];
      float g1 = acc1b[r], u1 = acc3b[r];
      // silu(g)*u with fast rcp (avoids the IEEE div_scale/newton chain)
      float h0 = g0 * __builtin_amdgcn_rcpf(1.0f + __expf(-g0)) * u0;
      float h1 = g1 * __builtin_amdgcn_rcpf(1.0f + __expf(-g1)) * u1;
      hs[m * DIM_H + col0 + ncol]        = f2bf(h0);
      hs[(m + 16) * DIM_H + col0 + ncol] = f2bf(h1);
    }
  }
  __syncthreads();

  // ---- Phase 2: y[32, 1024] = h @ w2; scale by routing weight; scatter-add ----
  const bf16_t* hrow0 = hs + mrow * DIM_H;
  const bf16_t* hrow1 = hs + (16 + mrow) * DIM_H;
  for (int j = 0; j < 8; ++j) {
    const int col0 = (wv * 8 + j) * 16;
    v8f acca = {}, accb = {};
    for (int kk = 0; kk < DIM_H / 32; ++kk) {
      const int k0 = kk * 32 + koff;
      v16bf a0 = lds_a_frag(hrow0 + kk * 32, koff);
      v16bf a1 = lds_a_frag(hrow1 + kk * 32, koff);
      v16bf b  = glb_b_frag(w2e, k0, col0 + ncol, DIM_D);
      acca = wmma_bf16(a0, b, acca);
      accb = wmma_bf16(a1, b, accb);
    }
    #pragma unroll
    for (int r = 0; r < 8; ++r) {
      const int m0 = r + rbase;
      const int m1 = m0 + 16;
      bool v0 = m0 < cnt_rem, v1 = m1 < cnt_rem;
      int  t0 = tokl[v0 ? m0 : 0],        t1 = tokl[v1 ? m1 : 0];
      float q0 = v0 ? wgtl[m0] : 0.0f,    q1 = v1 ? wgtl[m1] : 0.0f;
      atomicAdd(&out[(size_t)t0 * DIM_D + col0 + ncol], acca[r] * q0);
      atomicAdd(&out[(size_t)t1 * DIM_D + col0 + ncol], accb[r] * q1);
    }
  }
}

// ---------------- Aux loss: E * sum((usage/T)^2) ----------------
__global__ void aux_kernel(const float* __restrict__ usage, float* __restrict__ o) {
  if (threadIdx.x == 0) {
    float s = 0.f;
    for (int e = 0; e < NEXP; ++e) { float u = usage[e] * (1.0f / T_TOK); s += u * u; }
    o[0] = (float)NEXP * s;
  }
}

extern "C" void kernel_launch(void* const* d_in, const int* in_sizes, int n_in,
                              void* d_out, int out_size, void* d_ws, size_t ws_size,
                              hipStream_t stream) {
  const float* x  = (const float*)d_in[0];   // [T, D]
  const float* gw = (const float*)d_in[1];   // [D, E]
  const float* w1 = (const float*)d_in[2];   // [E, D, H]
  const float* w2 = (const float*)d_in[3];   // [E, H, D]  (dict order: w1, w2, w3)
  const float* w3 = (const float*)d_in[4];   // [E, D, H]
  float* out = (float*)d_out;                // [T*D] ++ [1] aux

  uint8_t* ws = (uint8_t*)d_ws;
  int*   counts = (int*)ws;                                                // 8 ints
  float* usage  = (float*)(ws + 64);                                       // 8 floats
  int*   tok    = (int*)(ws + 128);                                        // E*T ints (256 KB)
  float* wgt    = (float*)(ws + 128 + (size_t)NEXP * T_TOK * sizeof(int)); // E*T floats

  hipMemsetAsync(d_ws, 0, 128, stream);
  hipMemsetAsync(d_out, 0, (size_t)T_TOK * DIM_D * sizeof(float), stream);

  router_kernel<<<T_TOK / 8, 256, 0, stream>>>(x, gw, counts, tok, wgt, usage);
  moe_expert_kernel<<<NEXP * 256, 256, 0, stream>>>(x, w1, w3, w2, counts, tok, wgt, out);
  aux_kernel<<<1, 32, 0, stream>>>(usage, out + (size_t)T_TOK * DIM_D);
}
